// LSTM_3633542332542
// MI455X (gfx1250) — matrix operand order repaired and checked by reading, not compiled, not run on hardware
//
#include <hip/hip_runtime.h>

// Problem constants (match the reference).
#define TT 8192
#define DD 1024
#define HH 1024
#define GG 4096   // 4*H
#define FORGET_BIAS_F 1.0f

typedef __attribute__((ext_vector_type(16))) __bf16 v16bf;
typedef __attribute__((ext_vector_type(8)))  float  v8f;

union BF16x16 {
  v16bf v;
  unsigned short u[16];
  uint4 q[2];
};

__device__ __forceinline__ unsigned short f2bf(float f) {
  union { float f; unsigned u; } a;
  a.f = f;
  unsigned u = a.u;
  u += 0x7FFFu + ((u >> 16) & 1u);   // round-to-nearest-even
  return (unsigned short)(u >> 16);
}

__device__ __forceinline__ float sigmoidf_(float x) {
  return 1.0f / (1.0f + __expf(-x));
}

// ---------------------------------------------------------------------------
// Reset the grid-barrier counter (re-run every call so graph replay is clean).
// ---------------------------------------------------------------------------
__global__ __launch_bounds__(256) void zero_counter_kernel(unsigned* counter) {
  if (threadIdx.x == 0 && blockIdx.x == 0) *counter = 0u;
}

// ---------------------------------------------------------------------------
// Elementwise f32 -> bf16, 8 elements/thread, vectorized 128b loads/stores.
// ---------------------------------------------------------------------------
__global__ __launch_bounds__(256)
void convert_to_bf16(const float* __restrict__ src, unsigned short* __restrict__ dst,
                     size_t n) {
  size_t i = ((size_t)blockIdx.x * 256 + threadIdx.x) * 8;
  if (i + 8 <= n) {
    float4 a = *(const float4*)(src + i);
    float4 b = *(const float4*)(src + i + 4);
    unsigned short o[8];
    o[0] = f2bf(a.x); o[1] = f2bf(a.y); o[2] = f2bf(a.z); o[3] = f2bf(a.w);
    o[4] = f2bf(b.x); o[5] = f2bf(b.y); o[6] = f2bf(b.z); o[7] = f2bf(b.w);
    *(uint4*)(dst + i) = *(const uint4*)o;
  }
}

// ---------------------------------------------------------------------------
// dst[N][K] (bf16, row-major over N) = transpose of src[K][N] (f32 row-major).
// One-time weight prep so WMMA B-fragments are contiguous 32B loads.
// ---------------------------------------------------------------------------
__global__ __launch_bounds__(256)
void transpose_to_bf16(const float* __restrict__ src, unsigned short* __restrict__ dst,
                       int K, int N) {
  int n = blockIdx.x * 32 + (threadIdx.x & 31);
  int k = blockIdx.y * 8 + (threadIdx.x >> 5);
  if (n < N && k < K)
    dst[(size_t)n * K + k] = f2bf(src[(size_t)k * N + n]);
}

// ---------------------------------------------------------------------------
// C[M,N] (f32) = A[M,K] (bf16 row-major) * BT[N,K] (bf16) + bias[N]
// Block = 8 waves; wave w owns a 64x32 macro-tile (4 M-tiles x 2 N-tiles) at
// rows blockIdx.y*512 + w*64, cols blockIdx.x*32. Software-pipelined K loop:
// fragments for k+32 are loaded before the 8 WMMAs for k are issued, so the
// scheduler can overlap global_load_b128 clauses with the matrix pipe instead
// of draining loadcnt to 0 before every WMMA.
// A frag: lanes 0-15 row M=lane K 0..15;
// lanes 16-31 same rows, K 16..31.
// B frag: lane n = col n K 0..15 ; lane n+16 = col n K 16..31.
// ---------------------------------------------------------------------------
__global__ __launch_bounds__(256)
void gemm_bf16_wmma(const unsigned short* __restrict__ A,
                    const unsigned short* __restrict__ BT,
                    const float* __restrict__ bias, float* __restrict__ C,
                    int M, int N, int K) {
  const int lane = threadIdx.x & 31;
  const int wave = threadIdx.x >> 5;
  const int half = lane >> 4;    // 0: lanes 0-15, 1: lanes 16-31
  const int l    = lane & 15;
  const int n0 = blockIdx.x * 32;
  const int m0 = blockIdx.y * 512 + wave * 64;

  const unsigned short* bp0 = BT + (size_t)(n0 + l) * K + half * 16;
  const unsigned short* bp1 = BT + (size_t)(n0 + 16 + l) * K + half * 16;
  const unsigned short* ap0 = A + (size_t)(m0 + l) * K + half * 16;
  const size_t arowstride = (size_t)16 * K;

  BF16x16 cb[2], ca[4];
  // Prologue: fragments for k0 = 0.
  cb[0].q[0] = *(const uint4*)(bp0);      cb[0].q[1] = *(const uint4*)(bp0 + 8);
  cb[1].q[0] = *(const uint4*)(bp1);      cb[1].q[1] = *(const uint4*)(bp1 + 8);
#pragma unroll
  for (int mt = 0; mt < 4; ++mt) {
    const unsigned short* p = ap0 + (size_t)mt * arowstride;
    ca[mt].q[0] = *(const uint4*)(p);
    ca[mt].q[1] = *(const uint4*)(p + 8);
  }

  v8f acc[8] = {{}, {}, {}, {}, {}, {}, {}, {}};
  for (int k0 = 0; k0 < K - 32; k0 += 32) {
    // Load k0+32 fragments into the "next" set first ...
    BF16x16 nb[2], na[4];
    const int kn = k0 + 32;
    nb[0].q[0] = *(const uint4*)(bp0 + kn);      nb[0].q[1] = *(const uint4*)(bp0 + kn + 8);
    nb[1].q[0] = *(const uint4*)(bp1 + kn);      nb[1].q[1] = *(const uint4*)(bp1 + kn + 8);
#pragma unroll
    for (int mt = 0; mt < 4; ++mt) {
      const unsigned short* p = ap0 + (size_t)mt * arowstride + kn;
      na[mt].q[0] = *(const uint4*)(p);
      na[mt].q[1] = *(const uint4*)(p + 8);
    }
    // ... then issue the 8 WMMAs for k0 (no dependency on the new loads).
#pragma unroll
    for (int mt = 0; mt < 4; ++mt) {
      acc[mt]     = __builtin_amdgcn_wmma_f32_16x16x32_bf16(false, ca[mt].v, false, cb[0].v,
                                                            (short)0, acc[mt], false, false);
      acc[4 + mt] = __builtin_amdgcn_wmma_f32_16x16x32_bf16(false, ca[mt].v, false, cb[1].v,
                                                            (short)0, acc[4 + mt], false, false);
    }
    cb[0] = nb[0]; cb[1] = nb[1];
#pragma unroll
    for (int mt = 0; mt < 4; ++mt) ca[mt] = na[mt];
  }
  // Epilogue: last k-step.
#pragma unroll
  for (int mt = 0; mt < 4; ++mt) {
    acc[mt]     = __builtin_amdgcn_wmma_f32_16x16x32_bf16(false, ca[mt].v, false, cb[0].v,
                                                          (short)0, acc[mt], false, false);
    acc[4 + mt] = __builtin_amdgcn_wmma_f32_16x16x32_bf16(false, ca[mt].v, false, cb[1].v,
                                                          (short)0, acc[4 + mt], false, false);
  }

#pragma unroll
  for (int nt = 0; nt < 2; ++nt) {
    const float bv = bias[n0 + nt * 16 + l];
#pragma unroll
    for (int mt = 0; mt < 4; ++mt) {
#pragma unroll
      for (int r = 0; r < 8; ++r)
        C[(size_t)(m0 + mt * 16 + half * 8 + r) * N + (n0 + nt * 16 + l)] =
            acc[nt * 4 + mt][r] + bv;
    }
  }
}

// ---------------------------------------------------------------------------
// Persistent recurrent scan. Grid = 256 blocks; block b owns H-slice
// j in [4b, 4b+4) => 16 gate columns {g*1024 + j}. Its 16x1024 bf16 slice of
// Wh^T lives in LDS (32 KB) for the whole kernel — zero DRAM weight traffic
// per step. Each step: copy bf16 h into LDS (2 KB), each wave WMMAs its
// 128-wide K-chunk (4 chained wmma), partial sums reduced with ds_add_f32,
// gates in fp32, h published (bf16) to a global double buffer, then a
// grid-wide arrive/spin barrier on a monotonic counter.
// ---------------------------------------------------------------------------
__global__ __launch_bounds__(256)
void lstm_scan_persistent(const float* __restrict__ xz,            // [T,4H] f32
                          const unsigned short* __restrict__ WhT,  // [4H][H] bf16
                          const float* __restrict__ c_in,          // [H]
                          const float* __restrict__ h_in,          // [H]
                          unsigned short* __restrict__ hs,         // [T,H] bf16
                          unsigned short* __restrict__ hbuf,       // [2][H] bf16
                          float* __restrict__ tail,                // c_last[H], h_last[H]
                          unsigned* counter, int nblocks) {
  __shared__ __align__(16) unsigned short wlds[16][HH];  // 32 KB weight slice
  __shared__ __align__(16) unsigned short hlds[HH];      // 2 KB  h (bf16)
  __shared__ float zbuf[16];
  __shared__ float cbuf[4];

  const int tid  = threadIdx.x;
  const int lane = tid & 31;
  const int wave = tid >> 5;
  const int half = lane >> 4;
  const int l16  = lane & 15;
  const int jbase = blockIdx.x * 4;

  // Stage this block's 16 columns of Wh^T into LDS (once).
  for (int i = tid; i < 16 * HH; i += 256) {
    int lc = i >> 10, k = i & (HH - 1);
    int gcol = (lc >> 2) * HH + jbase + (lc & 3);
    wlds[lc][k] = WhT[(size_t)gcol * HH + k];
  }
  if (tid < 4) cbuf[tid] = c_in[jbase + tid];
  __syncthreads();

  for (int t = 0; t < TT; ++t) {
    // Stage h_t (bf16) and seed z with the precomputed xz row.
    if (t == 0) {
      for (int i = tid; i < HH; i += 256) hlds[i] = f2bf(h_in[i]);
    } else {
      ((uint2*)hlds)[tid] = ((const uint2*)(hbuf + (size_t)(t & 1) * HH))[tid];
    }
    if (tid < 16) {
      int gcol = (tid >> 2) * HH + jbase + (tid & 3);
      zbuf[tid] = xz[(size_t)t * GG + gcol];
    }
    __syncthreads();

    // Wave 'wave' covers K in [wave*128, wave*128+128): 4 chained WMMAs.
    v8f acc = {};
#pragma unroll
    for (int it = 0; it < 4; ++it) {
      const int kb = wave * 128 + it * 32;
      BF16x16 afr;
      afr.q[0] = make_uint4(0u, 0u, 0u, 0u);
      afr.q[1] = make_uint4(0u, 0u, 0u, 0u);
      if (l16 == 0) {  // only A row M=0 carries h (lanes 0 and 16)
#pragma unroll
        for (int i = 0; i < 8; ++i) afr.u[i]     = hlds[kb + half * 16 + i];
#pragma unroll
        for (int i = 0; i < 8; ++i) afr.u[8 + i] = hlds[kb + half * 16 + 8 + i];
      }
      BF16x16 bfr;
      const uint4* b4 = (const uint4*)&wlds[l16][kb + half * 16];
      bfr.q[0] = b4[0];
      bfr.q[1] = b4[1];
      acc = __builtin_amdgcn_wmma_f32_16x16x32_bf16(false, afr.v, false, bfr.v,
                                                    (short)0, acc, false, false);
    }
    // Row M=0 lives in VGPR0 of lanes 0..15 (col = lane). Reduce across waves.
    if (lane < 16) atomicAdd(&zbuf[lane], acc[0]);   // ds_add_f32
    __syncthreads();

    // Gate math in fp32 for the block's 4 H-elements.
    if (tid < 4) {
      float ig = zbuf[0 + tid], jg = zbuf[4 + tid];
      float fg = zbuf[8 + tid], og = zbuf[12 + tid];
      float c_old = cbuf[tid];
      float c_new = c_old * sigmoidf_(fg + FORGET_BIAS_F) + sigmoidf_(ig) * tanhf(jg);
      float h_new = tanhf(c_new) * sigmoidf_(og);
      cbuf[tid] = c_new;
      int j = jbase + tid;
      unsigned short hb = f2bf(h_new);
      hbuf[(size_t)((t + 1) & 1) * HH + j] = hb;
      hs[(size_t)t * HH + j] = hb;
      if (t == TT - 1) { tail[j] = c_new; tail[HH + j] = h_new; }
    }
    // Grid-wide barrier: release h writes, arrive, spin on monotonic target.
    __threadfence();
    __syncthreads();
    if (tid == 0) {
      __hip_atomic_fetch_add(counter, 1u, __ATOMIC_ACQ_REL, __HIP_MEMORY_SCOPE_AGENT);
      const unsigned target = (unsigned)(t + 1) * (unsigned)nblocks;
      while (__hip_atomic_load(counter, __ATOMIC_ACQUIRE, __HIP_MEMORY_SCOPE_AGENT) < target)
        __builtin_amdgcn_s_sleep(2);
    }
    __syncthreads();
  }
}

// ---------------------------------------------------------------------------
extern "C" void kernel_launch(void* const* d_in, const int* in_sizes, int n_in,
                              void* d_out, int out_size, void* d_ws, size_t ws_size,
                              hipStream_t stream) {
  (void)in_sizes; (void)n_in; (void)out_size; (void)ws_size;
  const float* x      = (const float*)d_in[0];
  const float* c_in   = (const float*)d_in[1];
  const float* h_in   = (const float*)d_in[2];
  const float* W_lstm = (const float*)d_in[3];   // (D+H, 4H) row-major
  const float* b_lstm = (const float*)d_in[4];
  const float* W_out  = (const float*)d_in[5];   // (H, H)
  const float* b_out  = (const float*)d_in[6];
  float* out = (float*)d_out;                    // [T*H output | H c_last | H h_last]

  char* ws = (char*)d_ws;
  size_t off = 0;
  auto take = [&](size_t bytes) -> void* {
    void* p = (void*)(ws + off);
    off = (off + bytes + 255) & ~(size_t)255;
    return p;
  };
  unsigned*       counter = (unsigned*)      take(256);
  float*          xz      = (float*)         take((size_t)TT * GG * sizeof(float));
  unsigned short* xbf     = (unsigned short*)take((size_t)TT * DD * sizeof(unsigned short));
  unsigned short* hs      = (unsigned short*)take((size_t)TT * HH * sizeof(unsigned short));
  unsigned short* hbuf    = (unsigned short*)take((size_t)2 * HH * sizeof(unsigned short));
  unsigned short* WxT     = (unsigned short*)take((size_t)GG * DD * sizeof(unsigned short));
  unsigned short* WhT     = (unsigned short*)take((size_t)GG * HH * sizeof(unsigned short));
  unsigned short* WoT     = (unsigned short*)take((size_t)HH * HH * sizeof(unsigned short));

  zero_counter_kernel<<<1, 256, 0, stream>>>(counter);

  // One-time prep: activations + weights to bf16 (weights transposed to [N][K]).
  convert_to_bf16<<<dim3((unsigned)((size_t)TT * DD / (8 * 256))), 256, 0, stream>>>(
      x, xbf, (size_t)TT * DD);
  transpose_to_bf16<<<dim3(GG / 32, DD / 8), 256, 0, stream>>>(W_lstm, WxT, DD, GG);
  transpose_to_bf16<<<dim3(GG / 32, HH / 8), 256, 0, stream>>>(W_lstm + (size_t)DD * GG, WhT, HH, GG);
  transpose_to_bf16<<<dim3(HH / 32, HH / 8), 256, 0, stream>>>(W_out, WoT, HH, HH);

  // Phase 1: xz = x @ Wx + b_lstm   (8192 x 1024 x 4096)
  gemm_bf16_wmma<<<dim3(GG / 32, TT / 512), 256, 0, stream>>>(xbf, WxT, b_lstm, xz, TT, GG, DD);

  // Phase 2: persistent recurrent scan (Wh resident in LDS across 256 WGs).
  lstm_scan_persistent<<<HH / 4, 256, 0, stream>>>(xz, WhT, c_in, h_in, hs, hbuf,
                                                   out + (size_t)TT * HH, counter, HH / 4);

  // Phase 3: output = hs @ W_out + b_out   (8192 x 1024 x 1024)
  gemm_bf16_wmma<<<dim3(HH / 32, TT / 512), 256, 0, stream>>>(hs, WoT, b_out, out, TT, HH, HH);
}
